// MyESN_39213051412583
// MI455X (gfx1250) — matrix operand order repaired and checked by reading, not compiled, not run on hardware
//
#include <hip/hip_runtime.h>
#include <math.h>
#include <stdint.h>

// ---- problem constants (match reference) ----
#define T_STEPS 25000
#define RES     1024
#define WASHOUT 250

// ---- kernel geometry ----
#define NWG          16                    // persistent workgroups (cluster-sized, one per WGP)
#define ROWS_PER_WG  (RES / NWG)           // 64 rows of U per WG
#define WAVES_PER_WG 4                     // each wave owns a 16-row block
#define THREADS      (WAVES_PER_WG * 32)   // 128 threads, wave32
#define KTILES       (RES / 32)            // 32 K-tiles of 32 (bf16 WMMA K)

// ---- LDS layout (dynamic shared) ----
#define SA_BYTES   (WAVES_PER_WG * KTILES * 32 * 8 * 4)  // 128 KB: U as bf16 A-fragments
#define SXB_OFF    SA_BYTES                               // 2048 B bf16 x + 32 B zero pad
#define SV_OFF     (SXB_OFF + 2048 + 32)
#define SRED_OFF   (SV_OFF + ROWS_PER_WG * 4)
#define SMEM_BYTES (SRED_OFF + THREADS * 4)

typedef __bf16        v16bf __attribute__((ext_vector_type(16)));
typedef float         v8f   __attribute__((ext_vector_type(8)));
typedef unsigned int  v4u   __attribute__((ext_vector_type(4)));
typedef int           v8i   __attribute__((ext_vector_type(8)));
typedef int           v4i   __attribute__((ext_vector_type(4)));

#if defined(__has_builtin)
#  if __has_builtin(__builtin_amdgcn_tensor_load_to_lds)
#    define HAVE_TDM 1
#  endif
#endif
#ifndef HAVE_TDM
#  define HAVE_TDM 0
#endif

// Grid-wide barrier: monotonic L2 atomic counter (correct under plain launch) +
// CDNA5 cluster split-barrier (HW fast path under cluster dispatch; NOP when ClusterID==0).
__device__ __forceinline__ void grid_barrier(int* bar, int target) {
    __threadfence();                           // release x stores to agent scope
    __syncthreads();
    if (threadIdx.x == 0) {
        __builtin_amdgcn_s_cluster_barrier();  // s_barrier_signal -3 ; s_barrier_wait -3
        atomicAdd(bar, 1);
        while (__hip_atomic_load(bar, __ATOMIC_RELAXED, __HIP_MEMORY_SCOPE_AGENT) < target) {
            __builtin_amdgcn_s_sleep(2);
        }
    }
    __syncthreads();
    __threadfence();                           // acquire: drop stale WGP$ lines
}

#if HAVE_TDM
// Issue one TDM descriptor: copy 2048 bytes (512 x 4B elements, 1 row) from global -> LDS.
__device__ __forceinline__ void tdm_load_x(const void* gsrc, uint32_t lds_off) {
    const unsigned long long ga = (unsigned long long)(uintptr_t)gsrc;
    v4u g0;
    g0[0] = 1u;                                            // count=1, user descriptor
    g0[1] = lds_off;                                       // lds_addr (bytes)
    g0[2] = (uint32_t)(ga & 0xffffffffu);                  // global_addr[31:0]
    g0[3] = (uint32_t)((ga >> 32) & 0x01ffffffu)           // global_addr[56:32]
          | (2u << 30);                                    // type = 2 ("image")
    v8i g1;
    g1[0] = 0x20000;                                       // data_size=2 -> 4-byte elems
    g1[1] = (int)(512u << 16);                             // tensor_dim0 = 512
    g1[2] = (int)(1u << 16);                               // tensor_dim1 = 1
    g1[3] = (int)(512u << 16);                             // tile_dim0 = 512
    g1[4] = 1;                                             // tile_dim1 = 1, tile_dim2 = 0
    g1[5] = 512;                                           // tensor_dim0_stride = 512
    g1[6] = 0;
    g1[7] = 0;
    v4i z4 = {0, 0, 0, 0};
#if defined(__clang_major__) && (__clang_major__ <= 22)
    __builtin_amdgcn_tensor_load_to_lds(g0, g1, z4, z4, 0);            // ROCm 7.2 5-arg form
#else
    v8i z8 = {0, 0, 0, 0, 0, 0, 0, 0};
    __builtin_amdgcn_tensor_load_to_lds(g0, g1, z4, z4, z8, 0);        // clang-23 6-arg form
#endif
    __builtin_amdgcn_s_wait_tensorcnt(0);
}
#endif

// A-fragment element->K mapping for 16-bit A 16x32: contiguous order k = 16*(lane/16) + e (matches B).
__device__ __forceinline__ int a_elem_to_k(int lane, int e) {
    if (lane < 16) return e;
    else           return 16 + e;
}

__global__ __launch_bounds__(THREADS, 1)
void esn_persistent_kernel(const float* __restrict__ ts,    // [T]
                           const float* __restrict__ x0,    // [RES]
                           const float* __restrict__ u,     // [RES,RES]
                           const float* __restrict__ v,     // [RES]
                           const float* __restrict__ w,     // [RES+1]
                           float*       __restrict__ out,   // [T-WASHOUT]
                           int*         __restrict__ bar,
                           float*       __restrict__ xb0,   // f32 state ping
                           float*       __restrict__ xb1,   // f32 state pong
                           __bf16*      __restrict__ xbf0,  // bf16 state ping
                           __bf16*      __restrict__ xbf1)  // bf16 state pong
{
    extern __shared__ char smem[];
    uint32_t* sA     = (uint32_t*)smem;
    char*     sxbB   = smem + SXB_OFF;            // bf16 x (2048B) + zero pad (32B)
    uint32_t* sxb32  = (uint32_t*)sxbB;
    float*    sv     = (float*)(smem + SV_OFF);
    float*    sred   = (float*)(smem + SRED_OFF);

    const int wg      = blockIdx.x;
    const int tid     = threadIdx.x;
    const int lane    = tid & 31;
    const int wave    = tid >> 5;
    const int rowBase = wg * ROWS_PER_WG + wave * 16;
    const int mrow    = lane & 15;

    // ---------------- one-time init ----------------
    // Swizzle this WG's 64x1024 U block into LDS in bf16 WMMA A-fragment order.
    for (int j = 0; j < KTILES; ++j) {
        const float* urow = u + (size_t)(rowBase + mrow) * RES + j * 32;
        if (j + 1 < KTILES) __builtin_prefetch(urow + 32, 0, 0);   // global_prefetch_b8
        uint32_t* dst = sA + (size_t)((wave * KTILES + j) * 32 + lane) * 8;
        #pragma unroll
        for (int d = 0; d < 8; ++d) {
            const int K0 = a_elem_to_k(lane, 2 * d);               // pairs are (K0, K0+1)
            const __bf16 b0 = (__bf16)urow[K0];
            const __bf16 b1 = (__bf16)urow[K0 + 1];
            dst[d] = (uint32_t)__builtin_bit_cast(unsigned short, b0) |
                     ((uint32_t)__builtin_bit_cast(unsigned short, b1) << 16);
        }
    }
    for (int i = tid; i < ROWS_PER_WG; i += THREADS) {
        const int r = wg * ROWS_PER_WG + i;
        sv[i]   = v[r];
        xb0[r]  = x0[r];                       // publish x_0 (own rows), f32 + bf16 mirror
        xbf0[r] = (__bf16)x0[r];
    }
    if (tid < 8) sxb32[512 + tid] = 0;         // 32B zero pad: B source for dead lanes
    __syncthreads();
    grid_barrier(bar, NWG * 1);                // epoch 1: x_0 visible everywhere

    // Per-lane B-fragment addressing, hoisted: lanes 0/16 read x slices, others read the pad.
    const bool live    = (lane & 15) == 0;
    const int  bOff0   = live ? ((lane >> 4) * 32) : 2048;   // bytes into sxb
    const int  bStride = live ? 64 : 0;                      // bytes per k-tile
    const char* aB     = smem + (size_t)(wave * KTILES * 32 + lane) * 32;
    const uint32_t sxbLds = (uint32_t)(uintptr_t)sxbB;       // LDS offset = low 32 bits of flat address (ISA §10.2)

    // ---------------- 25k sequential steps ----------------
    for (int t = 0; t < T_STEPS; ++t) {
        const float*  xin    = (t & 1) ? xb1  : xb0;   // x_t  (f32)
        float*        xout   = (t & 1) ? xb0  : xb1;   // x_{t+1} (f32)
        const __bf16* xinbf  = (t & 1) ? xbf1 : xbf0;  // x_t  (bf16)
        __bf16*       xoutbf = (t & 1) ? xbf0 : xbf1;  // x_{t+1} (bf16)

        // Stage bf16 x_t into LDS: TDM async copy (2 KB) or manual dword copy fallback.
#if HAVE_TDM
        if (wave == 0) tdm_load_x(xinbf, sxbLds);
#else
        {
            const uint32_t* src32 = (const uint32_t*)xinbf;
            for (int i = tid; i < RES / 2; i += THREADS) sxb32[i] = src32[i];
        }
#endif
        if (tid == 0) __builtin_prefetch(&ts[t + 8], 0, 1);
        __syncthreads();

        const float yt = ts[t];

        // y[16 rows] = U_block @ x via bf16 WMMA, x in B column N=0; dual accumulators.
        v8f acc0 = {}, acc1 = {};
        #pragma unroll 2
        for (int j = 0; j < KTILES; j += 2) {
            v16bf a0 = *(const v16bf*)(aB + (size_t)j * 1024);
            v16bf b0 = *(const v16bf*)(sxbB + bOff0 + j * bStride);
            acc0 = __builtin_amdgcn_wmma_f32_16x16x32_bf16(false, a0, false, b0,
                                                           (short)0, acc0, false, false);
            v16bf a1 = *(const v16bf*)(aB + (size_t)(j + 1) * 1024);
            v16bf b1 = *(const v16bf*)(sxbB + bOff0 + (j + 1) * bStride);
            acc1 = __builtin_amdgcn_wmma_f32_16x16x32_bf16(false, a1, false, b1,
                                                           (short)0, acc1, false, false);
        }

        // D column 0: lane 0 holds rows M=0..7 (VGPR g), lane 16 rows M=8..15.
        if (live) {
            const int rsub = (lane >> 4) * 8;
            #pragma unroll
            for (int g = 0; g < 8; ++g) {
                const int rloc  = wave * 16 + rsub + g;
                const int rglob = wg * ROWS_PER_WG + rloc;
                const float pre = acc0[g] + acc1[g] + sv[rloc] * yt;
                const float xn  = 0.1f * xin[rglob] + tanhf(pre);   // (1-LEAK)=0.1
                xout[rglob]   = xn;
                xoutbf[rglob] = (__bf16)xn;
            }
        }

        grid_barrier(bar, NWG * (t + 2));      // x_{t+1} visible everywhere

        // WG0 overlaps the readout dot with other WGs' next step (fixed order -> deterministic).
        if (wg == 0 && t >= WASHOUT) {
            float p = 0.0f;
            for (int i = tid; i < RES; i += THREADS) p += xout[i] * w[i];
            sred[tid] = p;
            __syncthreads();
            for (int s = THREADS / 2; s > 0; s >>= 1) {
                if (tid < s) sred[tid] += sred[tid + s];
                __syncthreads();
            }
            if (tid == 0) out[t - WASHOUT] = sred[0] + w[RES];      // '+w[RES]' = ones column
            __syncthreads();
        }
    }
}

extern "C" void kernel_launch(void* const* d_in, const int* in_sizes, int n_in,
                              void* d_out, int out_size, void* d_ws, size_t ws_size,
                              hipStream_t stream) {
    const float* ts = (const float*)d_in[0];   // time_series [25000]
    const float* x0 = (const float*)d_in[1];   // x0 [1024]
    const float* u  = (const float*)d_in[2];   // u  [1024*1024]
    const float* v  = (const float*)d_in[3];   // v  [1024]
    const float* w  = (const float*)d_in[4];   // w  [1025]
    float* out = (float*)d_out;                // [24750]

    char* ws    = (char*)d_ws;
    int*  bar   = (int*)ws;                                       // counter @ 0
    float*  xb0  = (float*)(ws + 256);                            // f32 state ping (4 KB)
    float*  xb1  = (float*)(ws + 256 + 4096);                     // f32 state pong (4 KB)
    __bf16* xbf0 = (__bf16*)(ws + 256 + 8192);                    // bf16 mirror ping (2 KB)
    __bf16* xbf1 = (__bf16*)(ws + 256 + 8192 + 2048);             // bf16 mirror pong (2 KB)

    hipMemsetAsync(ws, 0, 256, stream);                           // zero barrier state

    esn_persistent_kernel<<<NWG, THREADS, SMEM_BYTES, stream>>>(
        ts, x0, u, v, w, out, bar, xb0, xb1, xbf0, xbf1);
}
